// GATCell_29978871726167
// MI455X (gfx1250) — matrix operand order repaired and checked by reading, not compiled, not run on hardware
//
#include <hip/hip_runtime.h>
#include <hip/hip_bf16.h>
#include <stdint.h>

// ---------------- problem constants ----------------
#define N_NODES 50000
#define N_EDGES 800000
#define IN_DIM 128          // X_DIM + THETA_DIM
#define X_DIM 64
#define HEADS 4
#define HIDDEN 32
#define TOT_COLS 416        // Q(128) | K(128) | V(128) | skip(32)
#define N_TILES 26          // 416 / 16
#define K_CHUNKS 4          // 128 / 32
#define OUT_DIM 32

typedef __attribute__((ext_vector_type(8)))  float  v8f;
typedef __attribute__((ext_vector_type(16))) __bf16 v16bf;

// ---------------- helpers ----------------
__device__ __forceinline__ unsigned short f2bf(float f) {
    unsigned u = __float_as_uint(f);
    unsigned r = u + 0x7FFFu + ((u >> 16) & 1u);   // round-to-nearest-even
    return (unsigned short)(r >> 16);
}
// monotone float <-> uint mapping for atomicMax on floats
__device__ __forceinline__ unsigned fmap(float f) {
    unsigned u = __float_as_uint(f);
    return (u & 0x80000000u) ? ~u : (u | 0x80000000u);
}
__device__ __forceinline__ float funmap(unsigned u) {
    unsigned b = (u & 0x80000000u) ? (u & 0x7FFFFFFFu) : ~u;
    return __uint_as_float(b);
}

// ---------------- 1) pack weights into WMMA B-fragment layout (bf16 pairs) ----
// B 32x16 (KxN), this code's contiguous k order (same as A): lane = (K>=16 ? 16 : 0) + N ; VGPR g of the half holds
// the K pair 2g,2g+1. dword index = ((tile*4 + kchunk)*32 + lane)*8 + g
__global__ void pack_w_kernel(const float* __restrict__ Wq,
                              const float* __restrict__ Wk,
                              const float* __restrict__ Wv,
                              const float* __restrict__ Wskip,
                              uint32_t* __restrict__ pW) {
    int i = blockIdx.x * blockDim.x + threadIdx.x;
    if (i >= TOT_COLS * (IN_DIM / 2)) return;
    int c  = i / (IN_DIM / 2);     // output column 0..415
    int k  = (i % (IN_DIM / 2)) * 2;
    float f0, f1;
    if (c < 128)      { f0 = Wq[k*128 + c];          f1 = Wq[(k+1)*128 + c]; }
    else if (c < 256) { int cc = c-128; f0 = Wk[k*128 + cc];    f1 = Wk[(k+1)*128 + cc]; }
    else if (c < 384) { int cc = c-256; f0 = Wv[k*128 + cc];    f1 = Wv[(k+1)*128 + cc]; }
    else              { int cc = c-384; f0 = Wskip[k*32 + cc];  f1 = Wskip[(k+1)*32 + cc]; }
    int kc = k >> 5, kk = k & 31;
    int tile = c >> 4, n = c & 15;
    int lane = ((kk & 16) ? 16 : 0) + n;
    int g = (kk & 15) >> 1;
    pW[((tile * K_CHUNKS + kc) * 32 + lane) * 8 + g] =
        (uint32_t)f2bf(f0) | ((uint32_t)f2bf(f1) << 16);
}

// ---------------- 2) zero scratch ----------------
__global__ void zero_kernel(uint32_t* __restrict__ p, int n) {
    int i = blockIdx.x * blockDim.x + threadIdx.x;
    if (i < n) p[i] = 0u;
}

// ---------------- 3) fused projection GEMM: [N,128] x [128,416] ------------
// block = 256 threads (8 waves), one 16-row M tile per block.
__global__ __launch_bounds__(256)
void proj_gemm_kernel(const float* __restrict__ x, const float* __restrict__ theta,
                      const uint32_t* __restrict__ pW,
                      const float* __restrict__ bq, const float* __restrict__ bk,
                      const float* __restrict__ bv, const float* __restrict__ bskip,
                      float* __restrict__ Q, float* __restrict__ Ko,
                      float* __restrict__ V, float* __restrict__ S) {
    __shared__ unsigned short At[16][IN_DIM];   // 16x128 bf16 A tile (4 KB)

    const int m0  = blockIdx.x * 16;
    const int tid = threadIdx.x;

    // stage A tile (concat(x,theta) rows) as bf16 into LDS
#pragma unroll
    for (int i = 0; i < 8; ++i) {
        int idx = i * 256 + tid;            // 0..2047
        int row = idx >> 7;
        int col = idx & 127;
        int node = m0 + row;                // N_NODES % 16 == 0, always in range
        float v = (col < X_DIM) ? x[node * X_DIM + col]
                                : theta[node * X_DIM + (col - X_DIM)];
        At[row][col] = f2bf(v);
    }
    __syncthreads();

    // wave index as a proven-scalar: makes the tile loop, B base address and
    // the epilogue region select pure SALU (EXEC stays all-ones throughout).
    const int wv    = __builtin_amdgcn_readfirstlane(tid >> 5);
    const int lane  = tid & 31;
    const int lrow  = lane & 15;            // A: M row; B/C: N col
    const int lhalf = lane >> 4;            // 0 or 1
    const uint32_t* arow = reinterpret_cast<const uint32_t*>(&At[lrow][0]);

    union frag { uint32_t u[8]; v16bf v; };

    // load ALL A fragments once (invariant across tiles): 32 VGPRs
    frag afr[K_CHUNKS];
#pragma unroll
    for (int kc = 0; kc < K_CHUNKS; ++kc) {
        // lane<16 -> K {0..15}; lane>=16 -> K {16..31} (dword = K pair)
        int b0 = kc * 16 + (lhalf ? 8 : 0);
        int b1 = kc * 16 + (lhalf ? 12 : 4);
#pragma unroll
        for (int g = 0; g < 4; ++g) {
            afr[kc].u[g]     = arow[b0 + g];
            afr[kc].u[4 + g] = arow[b1 + g];
        }
    }

    for (int t = wv; t < N_TILES; t += 8) {      // t is scalar
        // all 4 B fragments for this tile: scalar base + lane offset,
        // immediate offsets select the K chunk (clauses of global_load_b128).
        const uint32_t* bp = pW + ((size_t)(t * K_CHUNKS) * 32 + lane) * 8;
        frag bfr[K_CHUNKS];
#pragma unroll
        for (int kc = 0; kc < K_CHUNKS; ++kc) {
#pragma unroll
            for (int g = 0; g < 8; ++g)
                bfr[kc].u[g] = bp[(size_t)kc * 32 * 8 + g];
        }

        v8f acc = {0.f, 0.f, 0.f, 0.f, 0.f, 0.f, 0.f, 0.f};
#pragma unroll
        for (int kc = 0; kc < K_CHUNKS; ++kc) {
            acc = __builtin_amdgcn_wmma_f32_16x16x32_bf16(
                false, afr[kc].v, false, bfr[kc].v, (short)0, acc, false, false);
        }

        // epilogue: region select on scalar c0 -> s_cmp/s_cbranch only.
        const int c0 = t * 16;
        const float* biasp; float* outp; int colbase; int stride;
        if (c0 < 128)      { biasp = bq;    outp = Q;  colbase = c0;       stride = 128; }
        else if (c0 < 256) { biasp = bk;    outp = Ko; colbase = c0 - 128; stride = 128; }
        else if (c0 < 384) { biasp = bv;    outp = V;  colbase = c0 - 256; stride = 128; }
        else               { biasp = bskip; outp = S;  colbase = c0 - 384; stride = 32;  }
        const int col  = colbase + lrow;    // per-lane column within region
        const float bias = biasp[col];
#pragma unroll
        for (int r = 0; r < 8; ++r) {
            int node = m0 + lhalf * 8 + r;  // C/D layout: lanes 0-15 M=r, 16-31 M=8+r
            outp[node * stride + col] = acc[r] + bias;
        }
    }
}

// ---------------- 4) edge logits + segment max ----------------
__global__ void edge_logits_kernel(const float* __restrict__ Q,
                                   const float* __restrict__ Kb,
                                   const int* __restrict__ ei,
                                   float* __restrict__ alpha,
                                   unsigned* __restrict__ amax) {
    int idx = blockIdx.x * blockDim.x + threadIdx.x;
    if (idx >= N_EDGES * HEADS) return;
    int e = idx >> 2, h = idx & 3;
    int s = ei[e];
    int d = ei[N_EDGES + e];
    const float4* qp = reinterpret_cast<const float4*>(Q  + (size_t)d * IN_DIM + h * HIDDEN);
    const float4* kp = reinterpret_cast<const float4*>(Kb + (size_t)s * IN_DIM + h * HIDDEN);
    float sum = 0.f;
#pragma unroll
    for (int i = 0; i < 8; ++i) {
        float4 a = qp[i], b = kp[i];
        sum += a.x*b.x + a.y*b.y + a.z*b.z + a.w*b.w;
    }
    sum *= 0.17677669529663689f;            // 1/sqrt(32)
    alpha[idx] = sum;
    atomicMax(&amax[d * HEADS + h], fmap(sum));
}

// ---------------- 5) exp + segment sum ----------------
__global__ void edge_exp_kernel(const int* __restrict__ ei,
                                float* __restrict__ alpha,
                                const unsigned* __restrict__ amax,
                                float* __restrict__ denom) {
    int idx = blockIdx.x * blockDim.x + threadIdx.x;
    if (idx >= N_EDGES * HEADS) return;
    int e = idx >> 2, h = idx & 3;
    int d = ei[N_EDGES + e];
    float ex = expf(alpha[idx] - funmap(amax[d * HEADS + h]));
    alpha[idx] = ex;
    atomicAdd(&denom[d * HEADS + h], ex);
}

// ---------------- 6) weighted scatter of V ----------------
__global__ void edge_accum_kernel(const int* __restrict__ ei,
                                  const float* __restrict__ alpha,
                                  const float* __restrict__ denom,
                                  const float* __restrict__ V,
                                  float* __restrict__ acc) {
    int idx = blockIdx.x * blockDim.x + threadIdx.x;
    if (idx >= N_EDGES * IN_DIM) return;
    int e = idx >> 7, j = idx & 127, h = j >> 5;
    int s = ei[e];
    int d = ei[N_EDGES + e];
    float attn = alpha[e * HEADS + h] / (denom[d * HEADS + h] + 1e-16f);
    atomicAdd(&acc[(size_t)d * IN_DIM + j], V[(size_t)s * IN_DIM + j] * attn);
}

// ---------------- 7) head-mean + skip + tanh + [32,32] matvec --------------
__global__ __launch_bounds__(256)
void finalize_kernel(const float* __restrict__ acc, const float* __restrict__ S,
                     const float* __restrict__ Wmlp, float* __restrict__ out) {
    __shared__ float hid[8][OUT_DIM];
    int tid = threadIdx.x;
    int ln = tid >> 5, o = tid & 31;
    int node = blockIdx.x * 8 + ln;         // N_NODES % 8 == 0
    const float* a = acc + (size_t)node * IN_DIM;
    float m = 0.25f * (a[o] + a[32 + o] + a[64 + o] + a[96 + o]) + S[node * HIDDEN + o];
    hid[ln][o] = tanhf(m);
    __syncthreads();
    float sum = 0.f;
#pragma unroll
    for (int c = 0; c < HIDDEN; ++c) sum += hid[ln][c] * Wmlp[c * OUT_DIM + o];
    out[(size_t)node * OUT_DIM + o] = sum;
}

// ---------------- launcher ----------------
extern "C" void kernel_launch(void* const* d_in, const int* in_sizes, int n_in,
                              void* d_out, int out_size, void* d_ws, size_t ws_size,
                              hipStream_t stream) {
    const float* x     = (const float*)d_in[0];
    const float* theta = (const float*)d_in[1];
    const int*   ei    = (const int*)  d_in[2];
    const float* Wq    = (const float*)d_in[3];
    const float* bq    = (const float*)d_in[4];
    const float* Wk    = (const float*)d_in[5];
    const float* bk    = (const float*)d_in[6];
    const float* Wv    = (const float*)d_in[7];
    const float* bv    = (const float*)d_in[8];
    const float* Wskip = (const float*)d_in[9];
    const float* bskip = (const float*)d_in[10];
    const float* Wmlp  = (const float*)d_in[11];
    float* out = (float*)d_out;

    // workspace layout (bytes, all 256-aligned)
    char* ws = (char*)d_ws;
    const size_t PW_B    = (size_t)N_TILES * K_CHUNKS * 32 * 8 * 4;   // 106,496
    const size_t QKV_B   = (size_t)N_NODES * IN_DIM * 4;              // 25.6 MB each
    const size_t SKIP_B  = (size_t)N_NODES * HIDDEN * 4;              // 6.4 MB
    const size_t ALPHA_B = (size_t)N_EDGES * HEADS * 4;               // 12.8 MB
    const size_t AMAX_B  = (size_t)N_NODES * HEADS * 4;
    const size_t DEN_B   = (size_t)N_NODES * HEADS * 4;
    const size_t ACC_B   = (size_t)N_NODES * IN_DIM * 4;

    uint32_t* pW   = (uint32_t*)(ws);
    float*    Qb   = (float*)(ws + PW_B);
    float*    Kbuf = (float*)(ws + PW_B + QKV_B);
    float*    Vb   = (float*)(ws + PW_B + 2 * QKV_B);
    float*    Sb   = (float*)(ws + PW_B + 3 * QKV_B);
    float*    al   = (float*)(ws + PW_B + 3 * QKV_B + SKIP_B);
    unsigned* amax = (unsigned*)(ws + PW_B + 3 * QKV_B + SKIP_B + ALPHA_B);
    float*    den  = (float*)(ws + PW_B + 3 * QKV_B + SKIP_B + ALPHA_B + AMAX_B);
    float*    accb = (float*)(ws + PW_B + 3 * QKV_B + SKIP_B + ALPHA_B + AMAX_B + DEN_B);

    // 1) pack weights (bf16, WMMA B-fragment order)
    {
        int n = TOT_COLS * (IN_DIM / 2);
        pack_w_kernel<<<(n + 255) / 256, 256, 0, stream>>>(Wq, Wk, Wv, Wskip, pW);
    }
    // 2) zero [amax | denom | acc] (contiguous)
    {
        int ndw = (int)((AMAX_B + DEN_B + ACC_B) / 4);
        zero_kernel<<<(ndw + 255) / 256, 256, 0, stream>>>((uint32_t*)amax, ndw);
    }
    // 3) fused QKV + skip projection (WMMA bf16, f32 accumulate)
    proj_gemm_kernel<<<N_NODES / 16, 256, 0, stream>>>(
        x, theta, pW, bq, bk, bv, bskip, Qb, Kbuf, Vb, Sb);
    // 4) per-edge logits + segment max
    {
        int n = N_EDGES * HEADS;
        edge_logits_kernel<<<(n + 255) / 256, 256, 0, stream>>>(Qb, Kbuf, ei, al, amax);
    }
    // 5) exp + segment sum
    {
        int n = N_EDGES * HEADS;
        edge_exp_kernel<<<(n + 255) / 256, 256, 0, stream>>>(ei, al, amax, den);
    }
    // 6) attention-weighted scatter of V
    {
        int n = N_EDGES * IN_DIM;
        edge_accum_kernel<<<(n + 255) / 256, 256, 0, stream>>>(ei, al, den, Vb, accb);
    }
    // 7) head mean + skip + tanh + output matvec
    finalize_kernel<<<N_NODES / 8, 256, 0, stream>>>(accb, Sb, Wmlp, out);
}
